// MultiHeadAttention_59064390254978
// MI455X (gfx1250) — matrix operand order repaired and checked by reading, not compiled, not run on hardware
//
#include <hip/hip_runtime.h>
#include <hip/hip_bf16.h>

// ---------------------------------------------------------------------------
// MHA forward for MI455X (gfx1250): f16 WMMA (v_wmma_f32_16x16x32_f16),
// flash-attention single pass, wave32 fragment layouts per CDNA5 ISA 7.12.2.
// B=2, S=2048, D=1024, H=16, hd=64.
// GEMMs use 32x64 per-wave tiles: 8 WMMA per 12 global_load_b128 to keep the
// matrix pipe (not L2 bandwidth) the limiter.
// ---------------------------------------------------------------------------

typedef __attribute__((ext_vector_type(16))) _Float16 v16h;
typedef __attribute__((ext_vector_type(8)))  _Float16 hvec8;
typedef __attribute__((ext_vector_type(8)))  float    v8f;

#define BATCH 2
#define SEQ   2048
#define DMODEL 1024
#define NHEAD 16
#define HDIM  64

__device__ __forceinline__ v8f wmma_f16(v16h a, v16h b, v8f c) {
  // (neg_a, A, neg_b, B, c_mod, C, reuse_a, reuse_b)
  return __builtin_amdgcn_wmma_f32_16x16x32_f16(false, a, false, b, (short)0, c,
                                                false, false);
}

// A-fragment (16x32, 16-bit) from row-major [*, ld] source.
// lane<16: halves0-7 = K kk+0..7,  halves8-15 = K kk+16..23  (row = lane)
// lane>=16: halves0-7 = K kk+8..15, halves8-15 = K kk+24..31 (row = lane-16)
__device__ __forceinline__ v16h frag_a_rm(const _Float16* p, int ld, int row0,
                                          int kk, int lane) {
  int r  = row0 + (lane & 15);
  int ko = kk + ((lane & 16) ? 8 : 0);
  const _Float16* s = p + (size_t)r * ld + ko;
  hvec8 lo = *(const hvec8*)(s);
  hvec8 hi = *(const hvec8*)(s + 16);
  v16h a;
#pragma unroll
  for (int i = 0; i < 8; ++i) { a[i] = lo[i]; a[i + 8] = hi[i]; }
  return a;
}

// B-fragment (32x16, 16-bit): lane owns column N = lane&15;
// lane<16: halves0-7 = K kk+0..7, halves8-15 = K kk+16..23; lane>=16: +8.
// colptr points at this lane's column data stored contiguously in K.
__device__ __forceinline__ v16h frag_b_cols(const _Float16* colptr, int kk,
                                            int lane) {
  int ko = kk + ((lane & 16) ? 8 : 0);
  hvec8 lo = *(const hvec8*)(colptr + ko);
  hvec8 hi = *(const hvec8*)(colptr + ko + 16);
  v16h b;
#pragma unroll
  for (int i = 0; i < 8; ++i) { b[i] = lo[i]; b[i + 8] = hi[i]; }
  return b;
}

// ---------------------------------------------------------------------------
__global__ void mha_cast_f16(const float* __restrict__ s,
                             _Float16* __restrict__ d, int n) {
  int i = blockIdx.x * blockDim.x + threadIdx.x;
  if (i < n) d[i] = (_Float16)s[i];
}

// s: rows x cols (row-major). d: cols x rows, i.e. d[n*rows + k] = s[k*cols + n]
__global__ void mha_transpose_f16(const float* __restrict__ s,
                                  _Float16* __restrict__ d, int rows, int cols) {
  int i = blockIdx.x * blockDim.x + threadIdx.x;
  if (i >= rows * cols) return;
  int n = i / rows, k = i - n * rows;
  d[i] = (_Float16)s[(size_t)k * cols + n];
}

// ---------------------------------------------------------------------------
// qkv = xh(4096x1024) @ W (1024x3072), W given transposed (3072x1024).
// One wave computes a 32(M) x 64(N) tile.  Scatter into q,k:(B,H,S,hd) and
// vT:(B,H,hd,S), all f16.
__global__ void __launch_bounds__(128)
mha_gemm_qkv(const _Float16* __restrict__ A, const _Float16* __restrict__ Bt,
             _Float16* __restrict__ q, _Float16* __restrict__ k,
             _Float16* __restrict__ v) {
  const int lane = threadIdx.x & 31;
  const int wave = threadIdx.x >> 5;
  const int m0 = blockIdx.y * 32;
  const int n0 = blockIdx.x * 256 + wave * 64;

  v8f c0[4], c1[4];
#pragma unroll
  for (int j = 0; j < 4; ++j) { c0[j] = 0.0f; c1[j] = 0.0f; }

  const _Float16* col[4];
#pragma unroll
  for (int j = 0; j < 4; ++j)
    col[j] = Bt + (size_t)(n0 + 16 * j + (lane & 15)) * DMODEL;

  for (int kk = 0; kk < DMODEL; kk += 32) {
    v16h a0 = frag_a_rm(A, DMODEL, m0,      kk, lane);
    v16h a1 = frag_a_rm(A, DMODEL, m0 + 16, kk, lane);
#pragma unroll
    for (int j = 0; j < 4; ++j) {
      v16h b = frag_b_cols(col[j], kk, lane);
      c0[j] = wmma_f16(a0, b, c0[j]);
      c1[j] = wmma_f16(a1, b, c1[j]);
    }
  }

  const int rsel = (lane & 16) ? 8 : 0;
#pragma unroll
  for (int g = 0; g < 2; ++g) {
#pragma unroll
    for (int j = 0; j < 4; ++j) {
      int n = n0 + 16 * j + (lane & 15);
      int which = n >> 10;           // 0=q 1=k 2=v
      int inner = n & 1023;
      int hh = inner >> 6, dd = inner & 63;
#pragma unroll
      for (int i = 0; i < 8; ++i) {
        int m = m0 + 16 * g + i + rsel;
        int bb = m >> 11, ss = m & 2047;
        _Float16 val = (_Float16)(g ? c1[j][i] : c0[j][i]);
        size_t bh = (size_t)(bb * NHEAD + hh);
        if (which == 0)      q[((bh * SEQ + ss) << 6) + dd] = val;
        else if (which == 1) k[((bh * SEQ + ss) << 6) + dd] = val;
        else                 v[((bh << 6) + dd) * SEQ + ss] = val;
      }
    }
  }
}

// ---------------------------------------------------------------------------
// Flash attention: one wave per 16-query tile, 32-key blocks, causal.
__global__ void __launch_bounds__(128)
mha_attn(const _Float16* __restrict__ q, const _Float16* __restrict__ k,
         const _Float16* __restrict__ vt, _Float16* __restrict__ o) {
  __shared__ __align__(16) _Float16 plds[4][16 * 32];
  const int lane = threadIdx.x & 31;
  const int wave = threadIdx.x >> 5;
  const int tile = blockIdx.x * 4 + wave;
  const int qt = tile & 127;
  const int h  = (tile >> 7) & 15;
  const int b  = tile >> 11;
  const int qr0 = qt << 4;

  const size_t bh = (size_t)(b * NHEAD + h);
  const _Float16* qb = q  + (bh * SEQ) * HDIM;
  const _Float16* kb = k  + (bh * SEQ) * HDIM;
  const _Float16* vb = vt + (bh * HDIM) * SEQ;
  _Float16* pl = plds[wave];

  const v16h qa0 = frag_a_rm(qb, HDIM, qr0, 0,  lane);
  const v16h qa1 = frag_a_rm(qb, HDIM, qr0, 32, lane);

  float mr[8], lr[8];
  v8f oacc[4];
#pragma unroll
  for (int j = 0; j < 4; ++j) oacc[j] = 0.0f;
#pragma unroll
  for (int i = 0; i < 8; ++i) { mr[i] = -3.0e38f; lr[i] = 0.0f; }

  const int rsel = (lane & 16) ? 8 : 0;
  const int rowbase = qr0 + rsel;
  const int nkb = (qr0 + 47) >> 5;

  for (int kbk = 0; kbk < nkb; ++kbk) {
    const int k0 = kbk << 5;
    // ---- S = Q @ K^T (16 x 32), head-dim K split 0..31 / 32..63 ----
    v8f s0 = 0.0f, s1 = 0.0f;
    {
      const _Float16* kr0 = kb + (size_t)(k0 + (lane & 15)) * HDIM;
      s0 = wmma_f16(qa0, frag_b_cols(kr0, 0,  lane), s0);
      s0 = wmma_f16(qa1, frag_b_cols(kr0, 32, lane), s0);
      const _Float16* kr1 = kb + (size_t)(k0 + 16 + (lane & 15)) * HDIM;
      s1 = wmma_f16(qa0, frag_b_cols(kr1, 0,  lane), s1);
      s1 = wmma_f16(qa1, frag_b_cols(kr1, 32, lane), s1);
    }
    // ---- online softmax (rows live in elements, cols across 16 lanes) ----
    const int key0 = k0 + (lane & 15);
    const int key1 = key0 + 16;
    float p0[8], p1[8], al[8];
#pragma unroll
    for (int i = 0; i < 8; ++i) {
      int row = rowbase + i;
      float a = (key0 > row) ? -3.0e38f : (float)s0[i] * 0.125f;
      float c = (key1 > row) ? -3.0e38f : (float)s1[i] * 0.125f;
      float t = fmaxf(a, c);
      t = fmaxf(t, __shfl_xor(t, 1, 32));
      t = fmaxf(t, __shfl_xor(t, 2, 32));
      t = fmaxf(t, __shfl_xor(t, 4, 32));
      t = fmaxf(t, __shfl_xor(t, 8, 32));
      float mn = fmaxf(mr[i], t);
      float alpha = __expf(mr[i] - mn);
      mr[i] = mn;
      float e0 = __expf(a - mn);
      float e1 = __expf(c - mn);
      p0[i] = e0; p1[i] = e1;
      float sum = e0 + e1;
      sum += __shfl_xor(sum, 1, 32);
      sum += __shfl_xor(sum, 2, 32);
      sum += __shfl_xor(sum, 4, 32);
      sum += __shfl_xor(sum, 8, 32);
      lr[i] = lr[i] * alpha + sum;
      al[i] = alpha;
    }
#pragma unroll
    for (int i = 0; i < 8; ++i) {
      oacc[0][i] *= al[i]; oacc[1][i] *= al[i];
      oacc[2][i] *= al[i]; oacc[3][i] *= al[i];
    }
    // ---- relayout P (C-layout f32) -> A-fragment f16 via per-wave LDS ----
    {
      const int ccol = lane & 15;
#pragma unroll
      for (int i = 0; i < 8; ++i) {
        pl[(rsel + i) * 32 + ccol]      = (_Float16)p0[i];
        pl[(rsel + i) * 32 + ccol + 16] = (_Float16)p1[i];
      }
    }
    asm volatile("s_wait_dscnt 0" ::: "memory");  // DS in-order per wave; fence compiler
    v16h pa;
    {
      int r  = lane & 15;
      int c0 = (lane & 16) ? 8 : 0;
      hvec8 lo = *(const hvec8*)(pl + r * 32 + c0);
      hvec8 hi = *(const hvec8*)(pl + r * 32 + c0 + 16);
#pragma unroll
      for (int i = 0; i < 8; ++i) { pa[i] = lo[i]; pa[i + 8] = hi[i]; }
    }
    asm volatile("" ::: "memory");
    // ---- O += P @ V  (V^T stored (hd,S): lane=dim col, halves=keys) ----
#pragma unroll
    for (int j = 0; j < 4; ++j) {
      const _Float16* vr = vb + (size_t)(16 * j + (lane & 15)) * SEQ;
      oacc[j] = wmma_f16(pa, frag_b_cols(vr, k0, lane), oacc[j]);
    }
  }

#pragma unroll
  for (int i = 0; i < 8; ++i) lr[i] = 1.0f / lr[i];
#pragma unroll
  for (int j = 0; j < 4; ++j) {
    int d = h * HDIM + 16 * j + (lane & 15);
#pragma unroll
    for (int i = 0; i < 8; ++i) {
      int srow = rowbase + i;
      o[(((size_t)(b * SEQ + srow)) << 10) + d] = (_Float16)(oacc[j][i] * lr[i]);
    }
  }
}

// ---------------------------------------------------------------------------
// out = attn(4096x1024,f16) @ w_out(1024x1024), w_out given transposed; f32
// out.  One wave computes a 32(M) x 64(N) tile.
__global__ void __launch_bounds__(128)
mha_gemm_out(const _Float16* __restrict__ A, const _Float16* __restrict__ Bt,
             float* __restrict__ out) {
  const int lane = threadIdx.x & 31;
  const int wave = threadIdx.x >> 5;
  const int m0 = blockIdx.y * 32;
  const int n0 = blockIdx.x * 256 + wave * 64;

  v8f c0[4], c1[4];
#pragma unroll
  for (int j = 0; j < 4; ++j) { c0[j] = 0.0f; c1[j] = 0.0f; }

  const _Float16* col[4];
#pragma unroll
  for (int j = 0; j < 4; ++j)
    col[j] = Bt + (size_t)(n0 + 16 * j + (lane & 15)) * DMODEL;

  for (int kk = 0; kk < DMODEL; kk += 32) {
    v16h a0 = frag_a_rm(A, DMODEL, m0,      kk, lane);
    v16h a1 = frag_a_rm(A, DMODEL, m0 + 16, kk, lane);
#pragma unroll
    for (int j = 0; j < 4; ++j) {
      v16h b = frag_b_cols(col[j], kk, lane);
      c0[j] = wmma_f16(a0, b, c0[j]);
      c1[j] = wmma_f16(a1, b, c1[j]);
    }
  }
  const int rsel = (lane & 16) ? 8 : 0;
#pragma unroll
  for (int g = 0; g < 2; ++g) {
#pragma unroll
    for (int j = 0; j < 4; ++j) {
      int n = n0 + 16 * j + (lane & 15);
#pragma unroll
      for (int i = 0; i < 8; ++i) {
        int m = m0 + 16 * g + i + rsel;
        out[(size_t)m * DMODEL + n] = g ? c1[j][i] : c0[j][i];
      }
    }
  }
}

// ---------------------------------------------------------------------------
extern "C" void kernel_launch(void* const* d_in, const int* in_sizes, int n_in,
                              void* d_out, int out_size, void* d_ws,
                              size_t ws_size, hipStream_t stream) {
  const float* x     = (const float*)d_in[0];   // (2,2048,1024)
  const float* w_qkv = (const float*)d_in[1];   // (1024,3072)
  const float* w_out = (const float*)d_in[2];   // (1024,1024)
  float* out = (float*)d_out;                   // (2,2048,1024) f32

  char* ws = (char*)d_ws;
  const size_t MB = 1ull << 20;
  _Float16* xh    = (_Float16*)(ws + 0 * MB);   // 8 MB  x as f16
  _Float16* wqkvT = (_Float16*)(ws + 8 * MB);   // 6 MB  (3072,1024)
  _Float16* woutT = (_Float16*)(ws + 14 * MB);  // 2 MB  (1024,1024)
  _Float16* qh    = (_Float16*)(ws + 16 * MB);  // 8 MB  (B,H,S,hd)
  _Float16* kh    = (_Float16*)(ws + 24 * MB);  // 8 MB  (B,H,S,hd)
  _Float16* vth   = (_Float16*)(ws + 32 * MB);  // 8 MB  (B,H,hd,S)
  _Float16* ah    = (_Float16*)(ws + 40 * MB);  // 8 MB  attn out (B,S,D)

  const int NX = BATCH * SEQ * DMODEL;          // 4194304
  mha_cast_f16<<<(NX + 255) / 256, 256, 0, stream>>>(x, xh, NX);
  mha_transpose_f16<<<(DMODEL * 3 * DMODEL + 255) / 256, 256, 0, stream>>>(
      w_qkv, wqkvT, DMODEL, 3 * DMODEL);
  mha_transpose_f16<<<(DMODEL * DMODEL + 255) / 256, 256, 0, stream>>>(
      w_out, woutT, DMODEL, DMODEL);

  // QKV GEMM: M=4096 (128 tiles of 32), N=3072; block covers 32x256
  mha_gemm_qkv<<<dim3(12, 128), 128, 0, stream>>>(xh, wqkvT, qh, kh, vth);

  // Attention: B*H*(S/16) = 4096 tiles, 4 waves/block
  mha_attn<<<1024, 128, 0, stream>>>(qh, kh, vth, ah);

  // Output projection: M=4096 (128 tiles of 32), N=1024
  mha_gemm_out<<<dim3(4, 128), 128, 0, stream>>>(ah, woutT, out);
}
